// EGNN_22050362097722
// MI455X (gfx1250) — matrix operand order repaired and checked by reading, not compiled, not run on hardware
//
#include <hip/hip_runtime.h>

// ---------------------------------------------------------------------------
// EGNN forward for MI455X (gfx1250, wave32, WMMA bf16 w/ fp32 accumulate)
// ---------------------------------------------------------------------------

typedef __attribute__((ext_vector_type(16))) __bf16 v16bf;
typedef __attribute__((ext_vector_type(8)))  float  v8f;

#define DEVINL __device__ __forceinline__

DEVINL unsigned short f2bfbits(float f) {
    unsigned u = __builtin_bit_cast(unsigned, f);
    u += 0x7FFFu + ((u >> 16) & 1u);           // round-to-nearest-even
    return (unsigned short)(u >> 16);
}
DEVINL __bf16 usbf(unsigned short s) { return __builtin_bit_cast(__bf16, s); }
DEVINL __bf16 f2bf(float f)          { return usbf(f2bfbits(f)); }
DEVINL float  bf2f(unsigned short s) { return __builtin_bit_cast(float, (unsigned)s << 16); }
DEVINL float  siluf(float v)         { return v / (1.0f + __expf(-v)); }

DEVINL v8f wmma_bf16(v16bf a, v16bf b, v8f c) {
    return __builtin_amdgcn_wmma_f32_16x16x32_bf16(false, a, false, b, (short)0, c,
                                                   false, false);
}

// A fragment (16x32 bf16). Lane l16 owns row M=l16; rowp points at that row's
// 64 bf16 features. half selects the K sub-slices per ISA layout.
DEVINL v16bf load_a_bf(const unsigned short* rowp, int k0, int half) {
    v16bf a;
    const unsigned short* p0 = rowp + k0 * 32 + half * 8;
    const unsigned short* p1 = p0 + 16;
#pragma unroll
    for (int i = 0; i < 8; ++i) { a[i] = usbf(p0[i]); a[i + 8] = usbf(p1[i]); }
    return a;
}
DEVINL v16bf load_a_f32(const float* rowp, int k0, int half) {
    v16bf a;
    const float* p0 = rowp + k0 * 32 + half * 8;
    const float* p1 = p0 + 16;
#pragma unroll
    for (int i = 0; i < 8; ++i) { a[i] = f2bf(p0[i]); a[i + 8] = f2bf(p1[i]); }
    return a;
}
// B fragment from LDS staged in fragment-major layout: 16 contiguous bf16/lane.
DEVINL v16bf load_b(const unsigned short* frag, int lane) {
    v16bf b;
    const unsigned short* p = frag + lane * 16;
#pragma unroll
    for (int i = 0; i < 16; ++i) b[i] = usbf(p[i]);
    return b;
}
// Stage a K x 64 fp32 weight matrix (row-major) into bf16 B-fragment layout:
// frag (k0,n0) element for (lane,i) = W[k0*32 + (lane>>4)*8 + (i&7) + (i>>3)*16][n0*16 + (lane&15)]
DEVINL void stage_w(const float* __restrict__ W, int K, unsigned short* dst) {
    int total = K * 64;
    for (int t = threadIdx.x; t < total; t += blockDim.x) {
        int fragid = t >> 9, within = t & 511;
        int lane = within >> 4, i = within & 15;
        int k0 = fragid >> 2, n0 = fragid & 3;
        int k = k0 * 32 + (lane >> 4) * 8 + (i & 7) + (i >> 3) * 16;
        int n = n0 * 16 + (lane & 15);
        dst[t] = f2bfbits(W[k * 64 + n]);
    }
}

// ---------------------------------------------------------------------------
// Kernel 1: input embedding + buffer init
// ---------------------------------------------------------------------------
__global__ void __launch_bounds__(256) egnn_init_kernel(
    const float* __restrict__ h_in, const float* __restrict__ x_in,
    const float* __restrict__ W, const float* __restrict__ B,
    float* __restrict__ h32, unsigned short* __restrict__ hbf,
    float* __restrict__ xb, float* __restrict__ xacc,
    float* __restrict__ agg, float* __restrict__ cnt, int N)
{
    int idx = blockIdx.x * blockDim.x + threadIdx.x;
    if (idx < N * 64) {
        int nd = idx >> 6, o = idx & 63;
        float a = B[o];
        const float* hr = h_in + (size_t)nd * 16;
#pragma unroll
        for (int k = 0; k < 16; ++k) a = fmaf(hr[k], W[k * 64 + o], a);
        h32[idx] = a;
        hbf[idx] = f2bfbits(a);
        agg[idx] = 0.f;
    }
    if (idx < N * 3) { xb[idx] = x_in[idx]; xacc[idx] = 0.f; }
    if (idx < N)     { cnt[idx] = 0.f; }
}

// in-degree of receivers
__global__ void __launch_bounds__(256) egnn_count_kernel(
    const int* __restrict__ row, float* __restrict__ cnt, int E)
{
    int e = blockIdx.x * blockDim.x + threadIdx.x;
    if (e < E) atomicAdd(&cnt[row[e]], 1.0f);
}

// ---------------------------------------------------------------------------
// Kernel 2: fused per-edge pipeline (one wave per 16-edge tile)
//   radial -> edge MLP (2 layers) -> coord MLP -> atomic scatter
// ---------------------------------------------------------------------------
__global__ void __launch_bounds__(256) egnn_edge_kernel(
    const unsigned short* __restrict__ hbf, const float* __restrict__ xb,
    const int* __restrict__ row, const int* __restrict__ col,
    const float* __restrict__ W1, const float* __restrict__ B1,
    const float* __restrict__ W2, const float* __restrict__ B2,
    const float* __restrict__ CW1, const float* __restrict__ CB1,
    const float* __restrict__ CW2,
    float* __restrict__ xacc, float* __restrict__ agg, int E)
{
    __shared__ unsigned short sW1[16 * 512];   // edge_w1 rows 0..127 (h_r | h_c)
    __shared__ unsigned short sW2[8 * 512];
    __shared__ unsigned short sCW1[8 * 512];
    __shared__ float sW1r[64], sB1[64], sB2[64], sCB1[64], sCW2[64];
    __shared__ unsigned short sM[8][1024];     // per-wave 16x64 bf16 staging
    __shared__ float sRad[8][16];
    __shared__ int   sRow[8][16];
    __shared__ float sPhi[8][16];

    stage_w(W1, 128, sW1);
    stage_w(W2, 64, sW2);
    stage_w(CW1, 64, sCW1);
    for (int t = threadIdx.x; t < 64; t += blockDim.x) {
        sW1r[t] = W1[128 * 64 + t];            // radial row of edge_w1
        sB1[t] = B1[t]; sB2[t] = B2[t]; sCB1[t] = CB1[t]; sCW2[t] = CW2[t];
    }
    __syncthreads();

    const int wave = threadIdx.x >> 5;
    const int lane = threadIdx.x & 31;
    const int half = lane >> 4, l16 = lane & 15;
    unsigned short* mt = sM[wave];
    const int ntiles = E >> 4;                 // E is a multiple of 16
    const int gstride = gridDim.x * 8;

    for (int tile = blockIdx.x * 8 + wave; tile < ntiles; tile += gstride) {
        int e = tile * 16 + l16;
        int r = row[e], c = col[e];
        float dx = xb[r * 3 + 0] - xb[c * 3 + 0];
        float dy = xb[r * 3 + 1] - xb[c * 3 + 1];
        float dz = xb[r * 3 + 2] - xb[c * 3 + 2];
        float rad = dx * dx + dy * dy + dz * dz;
        if (half == 0) { sRad[wave][l16] = rad; sRow[wave][l16] = r; }

        // A fragments straight from global bf16 h (lane = one h row)
        const unsigned short* hr = hbf + (size_t)r * 64;
        const unsigned short* hc = hbf + (size_t)c * 64;
        v16bf ar0 = load_a_bf(hr, 0, half), ar1 = load_a_bf(hr, 1, half);
        v16bf ac0 = load_a_bf(hc, 0, half), ac1 = load_a_bf(hc, 1, half);

        // ---- edge MLP layer 1: silu([h_r|h_c|rad] @ W1 + b1) ----
        v8f m[4];
#pragma unroll
        for (int n0 = 0; n0 < 4; ++n0) {
            int n = n0 * 16 + l16;
            float bias = sB1[n], w1r = sW1r[n];
            v8f acc;
#pragma unroll
            for (int i = 0; i < 8; ++i)
                acc[i] = fmaf(sRad[wave][half * 8 + i], w1r, bias);
            acc = wmma_bf16(ar0, load_b(sW1 + (0 * 4 + n0) * 512, lane), acc);
            acc = wmma_bf16(ar1, load_b(sW1 + (1 * 4 + n0) * 512, lane), acc);
            acc = wmma_bf16(ac0, load_b(sW1 + (2 * 4 + n0) * 512, lane), acc);
            acc = wmma_bf16(ac1, load_b(sW1 + (3 * 4 + n0) * 512, lane), acc);
#pragma unroll
            for (int i = 0; i < 8; ++i) acc[i] = siluf(acc[i]);
            m[n0] = acc;
        }
#pragma unroll
        for (int n0 = 0; n0 < 4; ++n0) {
            int n = n0 * 16 + l16;
#pragma unroll
            for (int i = 0; i < 8; ++i)
                mt[(half * 8 + i) * 64 + n] = f2bfbits(m[n0][i]);
        }

        // ---- edge MLP layer 2: m = silu(m @ W2 + b2) ----
        const unsigned short* mrow = mt + l16 * 64;
        {
            v16bf a0 = load_a_bf(mrow, 0, half), a1 = load_a_bf(mrow, 1, half);
#pragma unroll
            for (int n0 = 0; n0 < 4; ++n0) {
                int n = n0 * 16 + l16;
                v8f acc;
#pragma unroll
                for (int i = 0; i < 8; ++i) acc[i] = sB2[n];
                acc = wmma_bf16(a0, load_b(sW2 + (0 * 4 + n0) * 512, lane), acc);
                acc = wmma_bf16(a1, load_b(sW2 + (1 * 4 + n0) * 512, lane), acc);
#pragma unroll
                for (int i = 0; i < 8; ++i) acc[i] = siluf(acc[i]);
                m[n0] = acc;
            }
        }
#pragma unroll
        for (int n0 = 0; n0 < 4; ++n0) {        // overwrite staging with final m
            int n = n0 * 16 + l16;
#pragma unroll
            for (int i = 0; i < 8; ++i)
                mt[(half * 8 + i) * 64 + n] = f2bfbits(m[n0][i]);
        }

        // ---- coord MLP: phi = silu(m @ CW1 + cb1) @ cw2 ----
        float s[8];
#pragma unroll
        for (int i = 0; i < 8; ++i) s[i] = 0.f;
        {
            v16bf a0 = load_a_bf(mrow, 0, half), a1 = load_a_bf(mrow, 1, half);
#pragma unroll
            for (int n0 = 0; n0 < 4; ++n0) {
                int n = n0 * 16 + l16;
                v8f acc;
#pragma unroll
                for (int i = 0; i < 8; ++i) acc[i] = sCB1[n];
                acc = wmma_bf16(a0, load_b(sCW1 + (0 * 4 + n0) * 512, lane), acc);
                acc = wmma_bf16(a1, load_b(sCW1 + (1 * 4 + n0) * 512, lane), acc);
                float w = sCW2[n];
#pragma unroll
                for (int i = 0; i < 8; ++i) s[i] = fmaf(siluf(acc[i]), w, s[i]);
            }
        }
#pragma unroll
        for (int i = 0; i < 8; ++i) {           // reduce over N within 16-lane group
            float v = s[i];
            v += __shfl_xor(v, 8, 16);
            v += __shfl_xor(v, 4, 16);
            v += __shfl_xor(v, 2, 16);
            v += __shfl_xor(v, 1, 16);
            s[i] = v;
        }
        if (l16 == 0) {
#pragma unroll
            for (int i = 0; i < 8; ++i) sPhi[wave][half * 8 + i] = s[i];
        }
        if (half == 0) {                        // scatter trans = diff * phi
            float phi = sPhi[wave][l16];
            atomicAdd(&xacc[(size_t)r * 3 + 0], dx * phi);
            atomicAdd(&xacc[(size_t)r * 3 + 1], dy * phi);
            atomicAdd(&xacc[(size_t)r * 3 + 2], dz * phi);
        }
        // ---- scatter m into agg[row] (segment_sum) ----
#pragma unroll 4
        for (int ee = 0; ee < 16; ++ee) {
            int rr = sRow[wave][ee];
            float v0 = bf2f(mt[ee * 64 + lane * 2 + 0]);
            float v1 = bf2f(mt[ee * 64 + lane * 2 + 1]);
            atomicAdd(&agg[(size_t)rr * 64 + lane * 2 + 0], v0);
            atomicAdd(&agg[(size_t)rr * 64 + lane * 2 + 1], v1);
        }
    }
}

// ---------------------------------------------------------------------------
// Kernel 3: node MLP + residual + coordinate update (one wave per 16 nodes)
// ---------------------------------------------------------------------------
__global__ void __launch_bounds__(256) egnn_node_kernel(
    float* __restrict__ h32, unsigned short* __restrict__ hbf,
    float* __restrict__ xb, float* __restrict__ xacc,
    float* __restrict__ agg, const float* __restrict__ cnt,
    const float* __restrict__ W1, const float* __restrict__ B1,
    const float* __restrict__ W2, const float* __restrict__ B2, int N)
{
    __shared__ unsigned short sW1[16 * 512];   // node_w1: rows 0..63 h, 64..127 agg
    __shared__ unsigned short sW2[8 * 512];
    __shared__ float sB1[64], sB2[64];
    __shared__ unsigned short sH[8][1024];

    stage_w(W1, 128, sW1);
    stage_w(W2, 64, sW2);
    for (int t = threadIdx.x; t < 64; t += blockDim.x) { sB1[t] = B1[t]; sB2[t] = B2[t]; }
    __syncthreads();

    const int wave = threadIdx.x >> 5;
    const int lane = threadIdx.x & 31;
    const int half = lane >> 4, l16 = lane & 15;
    unsigned short* ht = sH[wave];
    const int ntiles = N >> 4;                 // N is a multiple of 16
    const int gstride = gridDim.x * 8;

    for (int tile = blockIdx.x * 8 + wave; tile < ntiles; tile += gstride) {
        int node = tile * 16 + l16;
        const unsigned short* hrow = hbf + (size_t)node * 64;
        const float* arow = agg + (size_t)node * 64;
        v16bf ah0 = load_a_bf(hrow, 0, half), ah1 = load_a_bf(hrow, 1, half);
        v16bf ag0 = load_a_f32(arow, 0, half), ag1 = load_a_f32(arow, 1, half);

        v8f hid[4];
#pragma unroll
        for (int n0 = 0; n0 < 4; ++n0) {
            int n = n0 * 16 + l16;
            v8f acc;
#pragma unroll
            for (int i = 0; i < 8; ++i) acc[i] = sB1[n];
            acc = wmma_bf16(ah0, load_b(sW1 + (0 * 4 + n0) * 512, lane), acc);
            acc = wmma_bf16(ah1, load_b(sW1 + (1 * 4 + n0) * 512, lane), acc);
            acc = wmma_bf16(ag0, load_b(sW1 + (2 * 4 + n0) * 512, lane), acc);
            acc = wmma_bf16(ag1, load_b(sW1 + (3 * 4 + n0) * 512, lane), acc);
#pragma unroll
            for (int i = 0; i < 8; ++i) acc[i] = siluf(acc[i]);
            hid[n0] = acc;
        }
#pragma unroll
        for (int n0 = 0; n0 < 4; ++n0) {
            int n = n0 * 16 + l16;
#pragma unroll
            for (int i = 0; i < 8; ++i)
                ht[(half * 8 + i) * 64 + n] = f2bfbits(hid[n0][i]);
        }
        const unsigned short* hr2 = ht + l16 * 64;
        v16bf b0 = load_a_bf(hr2, 0, half), b1 = load_a_bf(hr2, 1, half);
#pragma unroll
        for (int n0 = 0; n0 < 4; ++n0) {
            int n = n0 * 16 + l16;
            v8f acc;
#pragma unroll
            for (int i = 0; i < 8; ++i) acc[i] = sB2[n];
            acc = wmma_bf16(b0, load_b(sW2 + (0 * 4 + n0) * 512, lane), acc);
            acc = wmma_bf16(b1, load_b(sW2 + (1 * 4 + n0) * 512, lane), acc);
#pragma unroll
            for (int i = 0; i < 8; ++i) {      // residual, write both precisions
                int nd = tile * 16 + half * 8 + i;
                size_t idx = (size_t)nd * 64 + n;
                float nv = h32[idx] + acc[i];
                h32[idx] = nv;
                hbf[idx] = f2bfbits(nv);
            }
        }
        if (half == 0) {                       // x += xacc / max(cnt,1); reset
            int nd = tile * 16 + l16;
            float cc = fmaxf(cnt[nd], 1.0f);
#pragma unroll
            for (int d = 0; d < 3; ++d) {
                size_t ix = (size_t)nd * 3 + d;
                xb[ix] += xacc[ix] / cc;
                xacc[ix] = 0.f;
            }
        }
#pragma unroll 4
        for (int ee = 0; ee < 16; ++ee) {      // reset agg for next layer
            agg[(size_t)(tile * 16 + ee) * 64 + lane * 2 + 0] = 0.f;
            agg[(size_t)(tile * 16 + ee) * 64 + lane * 2 + 1] = 0.f;
        }
    }
}

// ---------------------------------------------------------------------------
// Kernel 4: output embedding + pack (h then x) into d_out
// ---------------------------------------------------------------------------
__global__ void __launch_bounds__(256) egnn_finish_kernel(
    const float* __restrict__ h32, const float* __restrict__ xb,
    const float* __restrict__ W, const float* __restrict__ B,
    float* __restrict__ out, int N)
{
    int idx = blockIdx.x * blockDim.x + threadIdx.x;
    if (idx < N * 64) {
        int nd = idx >> 6, o = idx & 63;
        float a = B[o];
        const float* hr = h32 + (size_t)nd * 64;
#pragma unroll 8
        for (int k = 0; k < 64; ++k) a = fmaf(hr[k], W[k * 64 + o], a);
        out[idx] = a;
    }
    if (idx < N * 3) out[(size_t)N * 64 + idx] = xb[idx];
}

// ---------------------------------------------------------------------------
extern "C" void kernel_launch(void* const* d_in, const int* in_sizes, int n_in,
                              void* d_out, int out_size, void* d_ws, size_t ws_size,
                              hipStream_t stream)
{
    const int N = in_sizes[0] / 16;
    const int E = in_sizes[2] / 2;
    const int L = 4;

    const float* h_in      = (const float*)d_in[0];
    const float* x_in      = (const float*)d_in[1];
    const int*   ei        = (const int*)d_in[2];
    const float* emb_in_w  = (const float*)d_in[3];
    const float* emb_in_b  = (const float*)d_in[4];
    const float* edge_w1   = (const float*)d_in[5];
    const float* edge_b1   = (const float*)d_in[6];
    const float* edge_w2   = (const float*)d_in[7];
    const float* edge_b2   = (const float*)d_in[8];
    const float* coord_w1  = (const float*)d_in[9];
    const float* coord_b1  = (const float*)d_in[10];
    const float* coord_w2  = (const float*)d_in[11];
    const float* node_w1   = (const float*)d_in[12];
    const float* node_b1   = (const float*)d_in[13];
    const float* node_w2   = (const float*)d_in[14];
    const float* node_b2   = (const float*)d_in[15];
    const float* emb_out_w = (const float*)d_in[16];
    const float* emb_out_b = (const float*)d_in[17];

    char* p = (char*)d_ws;
    auto take = [&](size_t bytes) {
        char* q = p;
        p += (bytes + 255) & ~(size_t)255;
        return q;
    };
    float*          h32  = (float*)take((size_t)N * 64 * 4);
    unsigned short* hbf  = (unsigned short*)take((size_t)N * 64 * 2);
    float*          xb   = (float*)take((size_t)N * 3 * 4);
    float*          xacc = (float*)take((size_t)N * 3 * 4);
    float*          agg  = (float*)take((size_t)N * 64 * 4);
    float*          cnt  = (float*)take((size_t)N * 4);
    (void)ws_size; (void)n_in; (void)out_size;

    const int thr = 256;
    egnn_init_kernel<<<(N * 64 + thr - 1) / thr, thr, 0, stream>>>(
        h_in, x_in, emb_in_w, emb_in_b, h32, hbf, xb, xacc, agg, cnt, N);
    egnn_count_kernel<<<(E + thr - 1) / thr, thr, 0, stream>>>(ei, cnt, E);

    const int etiles = E >> 4;
    const int eblocks = (etiles + 31) / 32;    // 4 tiles per wave
    const int ntiles = N >> 4;
    const int nblocks = (ntiles + 7) / 8;      // 1 tile per wave

    for (int l = 0; l < L; ++l) {
        egnn_edge_kernel<<<eblocks, thr, 0, stream>>>(
            hbf, xb, ei, ei + E,
            edge_w1 + (size_t)l * 129 * 64, edge_b1 + (size_t)l * 64,
            edge_w2 + (size_t)l * 64 * 64,  edge_b2 + (size_t)l * 64,
            coord_w1 + (size_t)l * 64 * 64, coord_b1 + (size_t)l * 64,
            coord_w2 + (size_t)l * 64,
            xacc, agg, E);
        egnn_node_kernel<<<nblocks, thr, 0, stream>>>(
            h32, hbf, xb, xacc, agg, cnt,
            node_w1 + (size_t)l * 128 * 64, node_b1 + (size_t)l * 64,
            node_w2 + (size_t)l * 64 * 64,  node_b2 + (size_t)l * 64, N);
    }

    egnn_finish_kernel<<<(N * 64 + thr - 1) / thr, thr, 0, stream>>>(
        h32, xb, emb_out_w, emb_out_b, (float*)d_out, N);
}
